// MHCN_62843961475851
// MI455X (gfx1250) — matrix operand order repaired and checked by reading, not compiled, not run on hardware
//
#include <hip/hip_runtime.h>
#include <cstdint>
#include <cstddef>

typedef __attribute__((ext_vector_type(16))) _Float16 v16h;
typedef __attribute__((ext_vector_type(8)))  _Float16 v8h;
typedef __attribute__((ext_vector_type(8)))  float    v8f;

#define DIMD 128
#define LDW  136   // LDS row stride in halfs: 272B = 17*16 -> 16B-aligned rows

// ---------------------------------------------------------------- utilities

__device__ __forceinline__ float waveSum(float v) {
  v += __shfl_xor(v, 16, 32);
  v += __shfl_xor(v, 8, 32);
  v += __shfl_xor(v, 4, 32);
  v += __shfl_xor(v, 2, 32);
  v += __shfl_xor(v, 1, 32);
  return v;
}

// single v_rcp_f32 (~1 ulp) instead of IEEE div expansion
__device__ __forceinline__ float fastRcp(float x) {
  return __builtin_amdgcn_rcpf(x);
}

__global__ void fill_zero_kernel(float* __restrict__ p, int n) {
  int i = blockIdx.x * blockDim.x + threadIdx.x;
  if (i < n) p[i] = 0.0f;
}

__global__ void copy_kernel(const float* __restrict__ src, float* __restrict__ dst, int n) {
  int i = blockIdx.x * blockDim.x + threadIdx.x;
  if (i < n) dst[i] = src[i];
}

// wa[e] = sum_d attention_mat[e][d] * attention[d]   (collapses chan-att GEMM)
__global__ void wa_kernel(const float* __restrict__ Wm, const float* __restrict__ a,
                          float* __restrict__ wa) {
  int e = threadIdx.x;  // 128 threads
  float s = 0.0f;
  #pragma unroll 8
  for (int d = 0; d < DIMD; ++d) s += Wm[e * DIMD + d] * a[d];
  wa[e] = s;
}

// ------------------------------------------------ gating GEMM via WMMA f16
// out = E * sigmoid(E @ W + b), dual-stored to out0 and out1.
// Block = 256 threads = 8 waves; wave w owns rows [blk*128 + w*16, +16).
// W is staged TRANSPOSED in LDS so each B fragment is two ds_load_b128.
__global__ __launch_bounds__(256) void gate_wmma_kernel(
    const float* __restrict__ E, const float* __restrict__ W,
    const float* __restrict__ bias,
    float* __restrict__ out0, float* __restrict__ out1, int nrows) {
  __shared__ _Float16 WhT[DIMD * LDW];   // WhT[n * LDW + k] = (f16) W[k][n]

  for (int i = threadIdx.x; i < DIMD * DIMD; i += 256) {
    int k = i >> 7, n = i & 127;         // coalesced global read of W[k][n]
    WhT[n * LDW + k] = (_Float16)W[i];
  }
  __syncthreads();

  const int wave = threadIdx.x >> 5;
  const int lane = threadIdx.x & 31;
  const int r0   = blockIdx.x * 128 + wave * 16;
  const int m16  = lane & 15;
  const int hi   = lane >> 4;     // half-wave selector
  const int abase = hi * 16;      // A: K base offset within 32-chunk
  const int mbase = hi * 8;       // C/D: M base

  // A fragments: 16x32 f16 per k-step (contiguous k order: k = hi*16 + s), loaded as two
  // contiguous 32B fp32 runs per lane, converted to f16.
  int arow = r0 + m16;
  if (arow >= nrows) arow = nrows - 1;   // branchless clamp keeps EXEC all-1s
  const float* rp = E + (size_t)arow * DIMD;
  v16h A[4];
  #pragma unroll
  for (int ks = 0; ks < 4; ++ks) {
    const float* p = rp + ks * 32 + abase;
    #pragma unroll
    for (int h = 0; h < 8; ++h) A[ks][h] = (_Float16)p[h];
    #pragma unroll
    for (int h = 0; h < 8; ++h) A[ks][8 + h] = (_Float16)p[8 + h];
  }

  #pragma unroll
  for (int nt = 0; nt < 8; ++nt) {
    const int n = nt * 16 + m16;
    const _Float16* brow = &WhT[n * LDW];
    v8f acc = {};
    #pragma unroll
    for (int ks = 0; ks < 4; ++ks) {
      const int koff = ks * 32 + hi * 16;
      // 16 contiguous halfs: K = koff .. koff+15 -> two ds_load_b128
      const v8h blo = *(const v8h*)(brow + koff);
      const v8h bhi = *(const v8h*)(brow + koff + 8);
      const v16h B = __builtin_shufflevector(blo, bhi,
          0, 1, 2, 3, 4, 5, 6, 7, 8, 9, 10, 11, 12, 13, 14, 15);
      acc = __builtin_amdgcn_wmma_f32_16x16x32_f16(
          /*neg_a=*/false, A[ks], /*neg_b=*/false, B,
          /*c_mod=*/(short)0, acc, /*reuse_a=*/false, /*reuse_b=*/false);
    }
    const float bn = bias[n];
    #pragma unroll
    for (int v = 0; v < 8; ++v) {
      const int row = r0 + mbase + v;
      if (row < nrows) {
        const float z = acc[v] + bn;
        const float e = E[(size_t)row * DIMD + n];
        const float g = e * fastRcp(1.0f + __expf(-z));   // v_exp + v_rcp
        out0[(size_t)row * DIMD + n] = g;
        out1[(size_t)row * DIMD + n] = g;
      }
    }
  }
}

// ------------------------------------------- channel attention mix (per row)
// out = sum_k softmax_k(dot(c_k, wa)) * c_k + 0.5 * cs
__global__ __launch_bounds__(256) void mix_kernel(
    const float* __restrict__ c0, const float* __restrict__ c1,
    const float* __restrict__ c2, const float* __restrict__ cs,
    const float* __restrict__ wa, float* __restrict__ out, int nrows) {
  const int wave = threadIdx.x >> 5, lane = threadIdx.x & 31;
  const int row = blockIdx.x * 8 + wave;
  if (row >= nrows) return;
  const size_t base = (size_t)row * DIMD + lane * 4;
  const float4 wv = *(const float4*)(wa + lane * 4);
  const float4 x0 = *(const float4*)(c0 + base);
  const float4 x1 = *(const float4*)(c1 + base);
  const float4 x2 = *(const float4*)(c2 + base);
  const float4 xs = *(const float4*)(cs + base);
  float w0 = waveSum(x0.x * wv.x + x0.y * wv.y + x0.z * wv.z + x0.w * wv.w);
  float w1 = waveSum(x1.x * wv.x + x1.y * wv.y + x1.z * wv.z + x1.w * wv.w);
  float w2 = waveSum(x2.x * wv.x + x2.y * wv.y + x2.z * wv.z + x2.w * wv.w);
  const float m = fmaxf(w0, fmaxf(w1, w2));
  const float e0 = __expf(w0 - m), e1 = __expf(w1 - m), e2 = __expf(w2 - m);
  const float inv = fastRcp(e0 + e1 + e2);
  const float s0 = e0 * inv, s1 = e1 * inv, s2 = e2 * inv;
  float4 o;
  o.x = s0 * x0.x + s1 * x1.x + s2 * x2.x + 0.5f * xs.x;
  o.y = s0 * x0.y + s1 * x1.y + s2 * x2.y + 0.5f * xs.y;
  o.z = s0 * x0.z + s1 * x1.z + s2 * x2.z + 0.5f * xs.z;
  o.w = s0 * x0.w + s1 * x1.w + s2 * x2.w + 0.5f * xs.w;
  *(float4*)(out + base) = o;
}

// ---------------------------------------------------- COO SpMM (one wave/edge)
// out[row[e], :] += val[e] * X[col[e], :]   (out must be pre-zeroed)
__global__ __launch_bounds__(256) void spmm_kernel(
    const int* __restrict__ rows, const int* __restrict__ cols,
    const float* __restrict__ vals, const float* __restrict__ X,
    float* __restrict__ out, int nE) {
  const int wave = threadIdx.x >> 5, lane = threadIdx.x & 31;
  const int e = blockIdx.x * 8 + wave;
  if (e >= nE) return;
  const int r = rows[e];
  const int c = cols[e];
  const float v = vals[e];
  const float4 x = *(const float4*)(X + (size_t)c * DIMD + lane * 4);
  float* o = out + (size_t)r * DIMD + lane * 4;
  unsafeAtomicAdd(o + 0, v * x.x);   // global_atomic_add_f32, resolves in L2
  unsafeAtomicAdd(o + 1, v * x.y);
  unsafeAtomicAdd(o + 2, v * x.z);
  unsafeAtomicAdd(o + 3, v * x.w);
}

// --------------------------------------------- acc += l2norm(X) (one wave/row)
__global__ __launch_bounds__(256) void l2acc_kernel(
    const float* __restrict__ X, float* __restrict__ acc, int nrows) {
  const int wave = threadIdx.x >> 5, lane = threadIdx.x & 31;
  const int row = blockIdx.x * 8 + wave;
  if (row >= nrows) return;
  const size_t base = (size_t)row * DIMD + lane * 4;
  const float4 x = *(const float4*)(X + base);
  const float ss = waveSum(x.x * x.x + x.y * x.y + x.z * x.z + x.w * x.w);
  const float r = rsqrtf(fmaxf(ss, 1e-12f));
  float4 a = *(float4*)(acc + base);
  a.x += x.x * r; a.y += x.y * r; a.z += x.z * r; a.w += x.w * r;
  *(float4*)(acc + base) = a;
}

// ---------------------------------------------------------------- host driver

extern "C" void kernel_launch(void* const* d_in, const int* in_sizes, int n_in,
                              void* d_out, int out_size, void* d_ws, size_t ws_size,
                              hipStream_t stream) {
  const float* u_emb = (const float*)d_in[0];
  const float* i_emb = (const float*)d_in[1];
  const float* gW    = (const float*)d_in[2];
  const float* gb    = (const float*)d_in[3];
  const float* att   = (const float*)d_in[4];
  const float* attM  = (const float*)d_in[5];
  const int*   Hrow[3] = {(const int*)d_in[6],  (const int*)d_in[9],  (const int*)d_in[12]};
  const int*   Hcol[3] = {(const int*)d_in[7],  (const int*)d_in[10], (const int*)d_in[13]};
  const float* Hval[3] = {(const float*)d_in[8],(const float*)d_in[11],(const float*)d_in[14]};
  const int*   Rrow = (const int*)d_in[15];
  const int*   Rcol = (const int*)d_in[16];
  const float* Rval = (const float*)d_in[17];

  const int D  = DIMD;
  const int NU = in_sizes[0] / D;
  const int NI = in_sizes[1] / D;
  const int EH = in_sizes[6];
  const int ER = in_sizes[15];
  const int NUD = NU * D, NID = NI * D;

  // workspace layout
  float* ws = (float*)d_ws;
  size_t off = 0;
  auto take = [&](size_t n) { float* p = ws + off; off += n; return p; };
  float* curA[3]; for (int k = 0; k < 3; ++k) curA[k] = take(NUD);
  float* curB[3]; for (int k = 0; k < 3; ++k) curB[k] = take(NUD);
  float* fc[3];   for (int k = 0; k < 3; ++k) fc[k]   = take(NUD);
  float* cur_sA = take(NUD);
  float* cur_sB = take(NUD);
  float* sum_s  = take(NUD);
  float* mixed  = take(NUD);
  float* cur_iA = take(NID);
  float* cur_iB = take(NID);
  float* wa     = take(D);
  (void)ws_size;

  float* outUser = (float*)d_out;
  float* outItem = (float*)d_out + (size_t)NUD;

  const dim3 B256(256);
  const dim3 Ggate((NU + 127) / 128);
  const dim3 GrowU((NU + 7) / 8);
  const dim3 GrowI((NI + 7) / 8);
  const dim3 GfillU((NUD + 255) / 256);
  const dim3 GfillI((NID + 255) / 256);
  const dim3 GspH((EH + 7) / 8);
  const dim3 GspR((ER + 7) / 8);

  // attention collapse vector
  wa_kernel<<<1, 128, 0, stream>>>(attM, att, wa);

  // gated initial embeddings; dual-store into running sums
  for (int c = 0; c < 3; ++c)
    gate_wmma_kernel<<<Ggate, B256, 0, stream>>>(u_emb, gW + (size_t)c * D * D,
                                                 gb + (size_t)c * D, curA[c], fc[c], NU);
  gate_wmma_kernel<<<Ggate, B256, 0, stream>>>(u_emb, gW + (size_t)3 * D * D,
                                               gb + (size_t)3 * D, cur_sA, sum_s, NU);

  // item accumulator starts at i_emb
  copy_kernel<<<GfillI, B256, 0, stream>>>(i_emb, outItem, NID);

  float* cur_cur[3] = {curA[0], curA[1], curA[2]};
  float* cur_nxt[3] = {curB[0], curB[1], curB[2]};
  float* cs_cur = cur_sA;
  float* cs_nxt = cur_sB;
  const float* ci_cur = i_emb;
  float* ibuf[2] = {cur_iA, cur_iB};

  for (int layer = 0; layer < 2; ++layer) {
    // mixed = chan_att(cur) + cur_s / 2
    mix_kernel<<<GrowU, B256, 0, stream>>>(cur_cur[0], cur_cur[1], cur_cur[2],
                                           cs_cur, wa, mixed, NU);
    // per-channel hypergraph propagation
    for (int k = 0; k < 3; ++k) {
      fill_zero_kernel<<<GfillU, B256, 0, stream>>>(cur_nxt[k], NUD);
      spmm_kernel<<<GspH, B256, 0, stream>>>(Hrow[k], Hcol[k], Hval[k],
                                             cur_cur[k], cur_nxt[k], EH);
      l2acc_kernel<<<GrowU, B256, 0, stream>>>(cur_nxt[k], fc[k], NU);
    }
    // new_item = R^T @ mixed
    float* ci_nxt = ibuf[layer];
    fill_zero_kernel<<<GfillI, B256, 0, stream>>>(ci_nxt, NID);
    spmm_kernel<<<GspR, B256, 0, stream>>>(Rcol, Rrow, Rval, mixed, ci_nxt, ER);
    l2acc_kernel<<<GrowI, B256, 0, stream>>>(ci_nxt, outItem, NI);
    // cur_s = R @ cur_i   (old cur_i)
    fill_zero_kernel<<<GfillU, B256, 0, stream>>>(cs_nxt, NUD);
    spmm_kernel<<<GspR, B256, 0, stream>>>(Rrow, Rcol, Rval, ci_cur, cs_nxt, ER);
    l2acc_kernel<<<GrowU, B256, 0, stream>>>(cs_nxt, sum_s, NU);
    // rotate buffers
    for (int k = 0; k < 3; ++k) { float* t = cur_cur[k]; cur_cur[k] = cur_nxt[k]; cur_nxt[k] = t; }
    { float* t = cs_cur; cs_cur = cs_nxt; cs_nxt = t; }
    ci_cur = ci_nxt;
  }

  // final_user = chan_att(fc) + sum(all_s)/2
  mix_kernel<<<GrowU, B256, 0, stream>>>(fc[0], fc[1], fc[2], sum_s, wa, outUser, NU);
  // final_item already accumulated in outItem
}
